// Attention_58368605552857
// MI455X (gfx1250) — matrix operand order repaired and checked by reading, not compiled, not run on hardware
//
#include <hip/hip_runtime.h>
#include <hip/hip_bf16.h>

typedef _Float16 half_t;
typedef __attribute__((ext_vector_type(16))) _Float16 v16h;
typedef __attribute__((ext_vector_type(8)))  _Float16 v8h;
typedef __attribute__((ext_vector_type(4)))  _Float16 v4h;
typedef __attribute__((ext_vector_type(8)))  float    v8f;

#define B_     4
#define N_     2048
#define DIM_   1024
#define HEADS_ 16
#define DH_    64
#define INNER_ 1024

union AV { v16h v; v8h h8[2]; };

static __device__ __forceinline__ v8f wmma_f16(v16h a, v16h b, v8f c) {
    // (neg_a, A, neg_b, B, c_mod, C, reuse_a, reuse_b)
    return __builtin_amdgcn_wmma_f32_16x16x32_f16(false, a, false, b, (short)0, c,
                                                  false, false);
}

// ---------------------------------------------------------------------------
// Kernel 1: qkv projections.  Y[b,n,e] = sum_c X[b,n,c] * W[e,c]
// grid (N/64, INNER/64, B*3), block 128 (4 waves). K-step 64 (8 WMMA/barrier).
// Output f16 in [B, H, N, 64] layout; q pre-scaled by DH^-0.5.
// ---------------------------------------------------------------------------
__global__ __launch_bounds__(128)
void proj_qkv_kernel(const float* __restrict__ q, const float* __restrict__ k,
                     const float* __restrict__ v,
                     const float* __restrict__ Wq, const float* __restrict__ Wk,
                     const float* __restrict__ Wv,
                     half_t* __restrict__ qh, half_t* __restrict__ kh,
                     half_t* __restrict__ vh) {
    __shared__ __attribute__((aligned(32))) half_t At[64 * 64];  // [m][k]
    __shared__ __attribute__((aligned(32))) half_t Bt[64 * 64];  // [n][k]

    const int t    = threadIdx.x;
    const int wave = t >> 5, lane = t & 31, hi = lane >> 4, lm = lane & 15;
    const int z = blockIdx.z;
    const int b = z / 3, which = z % 3;
    const int nBase = blockIdx.x * 64;
    const int eBase = blockIdx.y * 64;

    const float* X = (which == 0) ? q  : (which == 1) ? k  : v;
    const float* W = (which == 0) ? Wq : (which == 1) ? Wk : Wv;
    half_t*      Y = (which == 0) ? qh : (which == 1) ? kh : vh;
    const float scale = (which == 0) ? 0.125f : 1.0f;  // 64^-0.5

    v8f acc[4];
#pragma unroll
    for (int s = 0; s < 4; ++s) acc[s] = {};

    for (int kb = 0; kb < DIM_; kb += 64) {
        __syncthreads();
#pragma unroll
        for (int i = 0; i < 8; ++i) {
            const int idx = t + i * 128;            // 0..1023
            const int row = idx >> 4;               // 0..63
            const int c4  = (idx & 15) * 4;         // 0..60
            const float4 xa =
                *(const float4*)(X + ((size_t)b * N_ + nBase + row) * DIM_ + kb + c4);
            v4h ha; ha[0] = (half_t)xa.x; ha[1] = (half_t)xa.y;
                    ha[2] = (half_t)xa.z; ha[3] = (half_t)xa.w;
            *(v4h*)&At[row * 64 + c4] = ha;
            const float4 wb =
                *(const float4*)(W + (size_t)(eBase + row) * DIM_ + kb + c4);
            v4h hb; hb[0] = (half_t)wb.x; hb[1] = (half_t)wb.y;
                    hb[2] = (half_t)wb.z; hb[3] = (half_t)wb.w;
            *(v4h*)&Bt[row * 64 + c4] = hb;
        }
        __syncthreads();

        AV a0, a1;
        a0.h8[0] = *(const v8h*)&At[(wave * 16 + lm) * 64 + hi * 16];
        a0.h8[1] = *(const v8h*)&At[(wave * 16 + lm) * 64 + 8 + hi * 16];
        a1.h8[0] = *(const v8h*)&At[(wave * 16 + lm) * 64 + 32 + hi * 16];
        a1.h8[1] = *(const v8h*)&At[(wave * 16 + lm) * 64 + 40 + hi * 16];
#pragma unroll
        for (int sub = 0; sub < 4; ++sub) {
            const v16h bf0 = *(const v16h*)&Bt[(sub * 16 + lm) * 64 + hi * 16];
            const v16h bf1 = *(const v16h*)&Bt[(sub * 16 + lm) * 64 + 32 + hi * 16];
            acc[sub] = wmma_f16(a0.v, bf0, acc[sub]);
            acc[sub] = wmma_f16(a1.v, bf1, acc[sub]);
        }
    }

#pragma unroll
    for (int sub = 0; sub < 4; ++sub) {
#pragma unroll
        for (int r = 0; r < 8; ++r) {
            const int m  = nBase + wave * 16 + hi * 8 + r;
            const int e  = eBase + sub * 16 + lm;
            const int hh = e >> 6, d = e & 63;
            Y[(((size_t)b * HEADS_ + hh) * N_ + m) * DH_ + d] =
                (half_t)(acc[sub][r] * scale);
        }
    }
}

// ---------------------------------------------------------------------------
// Kernel 2: flash attention with rel-pos bias, 64-key window per step.
// grid (N/64, HEADS, B), block 128 (4 waves; each wave owns 16 query rows).
// Bias is streamed once with one-iteration-ahead prefetch.
// ---------------------------------------------------------------------------
__global__ __launch_bounds__(128)
void flash_attn_kernel(const half_t* __restrict__ qh, const half_t* __restrict__ kh,
                       const half_t* __restrict__ vh, const float* __restrict__ bias,
                       half_t* __restrict__ attO) {
    __shared__ __attribute__((aligned(32))) half_t Vt[64 * 64];     // [d][j]
    __shared__ __attribute__((aligned(32))) half_t Pt[4][16 * 64];  // per-wave [m][j]

    const int t    = threadIdx.x;
    const int wave = t >> 5, lane = t & 31, hi = lane >> 4, lm = lane & 15;
    const int mBase = blockIdx.x * 64;
    const int h = blockIdx.y;
    const int b = blockIdx.z;

    const size_t headOff = ((size_t)b * HEADS_ + h) * N_;
    const int qrow = mBase + wave * 16 + lm;
    const half_t* qp = qh + (headOff + qrow) * DH_;

    AV aq0, aq1;                                  // Q fragments (d = 0..31, 32..63)
    aq0.h8[0] = *(const v8h*)(qp + hi * 16);
    aq0.h8[1] = *(const v8h*)(qp + 8 + hi * 16);
    aq1.h8[0] = *(const v8h*)(qp + 32 + hi * 16);
    aq1.h8[1] = *(const v8h*)(qp + 40 + hi * 16);

    v8f o[4];
#pragma unroll
    for (int s = 0; s < 4; ++s) o[s] = {};
    float mrow[8], lrow[8];
#pragma unroll
    for (int r = 0; r < 8; ++r) { mrow[r] = -1e30f; lrow[r] = 0.0f; }

    const float* bp = bias + (size_t)h * N_ * N_;

    for (int jb = 0; jb < N_; jb += 64) {
        // Prefetch next bias window (dominant HBM stream): one 128B granule/lane.
        if (jb + 64 < N_) {
            const int prow = mBase + wave * 16 + (lane & 15);
            __builtin_prefetch(bp + (size_t)prow * N_ + jb + 64 + (lane >> 4) * 32, 0, 1);
        }
        __syncthreads();
        // Stage V block transposed: Vt[d][j] = vh[jb+j][d]
#pragma unroll
        for (int i = 0; i < 4; ++i) {
            const int u  = t + i * 128;           // 0..511
            const int jr = u >> 3;                // 0..63
            const int dc = (u & 7) * 8;           // 0..56
            const v8h tv = *(const v8h*)(vh + (headOff + jb + jr) * DH_ + dc);
#pragma unroll
            for (int qq = 0; qq < 8; ++qq) Vt[(dc + qq) * 64 + jr] = tv[qq];
        }
        __syncthreads();

        // S = Q K^T for this wave's 16 rows x 64 keys (four 16x16 C tiles)
        v8f s[4];
#pragma unroll
        for (int jsub = 0; jsub < 4; ++jsub) {
            const half_t* kp = kh + (headOff + jb + jsub * 16 + lm) * DH_;
            const v16h bk0 = *(const v16h*)(kp + hi * 16);
            const v16h bk1 = *(const v16h*)(kp + 32 + hi * 16);
            v8f z = {};
            z = wmma_f16(aq0.v, bk0, z);
            z = wmma_f16(aq1.v, bk1, z);
            s[jsub] = z;
        }
        // + rel_pos_bias (q already scaled)
#pragma unroll
        for (int jsub = 0; jsub < 4; ++jsub) {
#pragma unroll
            for (int r = 0; r < 8; ++r) {
                const int mg = mBase + wave * 16 + hi * 8 + r;
                s[jsub][r] += bp[(size_t)mg * N_ + jb + jsub * 16 + lm];
            }
        }
        // One online-softmax pass per 64-key window (row = 16 lanes of half-wave)
#pragma unroll
        for (int r = 0; r < 8; ++r) {
            float sm = fmaxf(fmaxf(s[0][r], s[1][r]), fmaxf(s[2][r], s[3][r]));
#pragma unroll
            for (int off = 1; off < 16; off <<= 1)
                sm = fmaxf(sm, __shfl_xor(sm, off, 16));
            const float mnew  = fmaxf(mrow[r], sm);
            const float alpha = __expf(mrow[r] - mnew);
            float pv[4], ps = 0.0f;
#pragma unroll
            for (int js = 0; js < 4; ++js) {
                pv[js] = __expf(s[js][r] - mnew);
                ps += pv[js];
            }
#pragma unroll
            for (int off = 1; off < 16; off <<= 1)
                ps += __shfl_xor(ps, off, 16);
            lrow[r] = lrow[r] * alpha + ps;
            mrow[r] = mnew;
#pragma unroll
            for (int f = 0; f < 4; ++f) o[f][r] *= alpha;
#pragma unroll
            for (int js = 0; js < 4; ++js)
                Pt[wave][(hi * 8 + r) * 64 + js * 16 + lm] = (half_t)pv[js];
        }
        // Re-read P as two A fragments (16x32 each), then O += P x V
        AV ap0, ap1;
        ap0.h8[0] = *(const v8h*)&Pt[wave][lm * 64 + hi * 16];
        ap0.h8[1] = *(const v8h*)&Pt[wave][lm * 64 + 8 + hi * 16];
        ap1.h8[0] = *(const v8h*)&Pt[wave][lm * 64 + 32 + hi * 16];
        ap1.h8[1] = *(const v8h*)&Pt[wave][lm * 64 + 40 + hi * 16];
#pragma unroll
        for (int sub = 0; sub < 4; ++sub) {
            const v16h bv0 = *(const v16h*)&Vt[(sub * 16 + lm) * 64 + hi * 16];
            const v16h bv1 = *(const v16h*)&Vt[(sub * 16 + lm) * 64 + 32 + hi * 16];
            o[sub] = wmma_f16(ap0.v, bv0, o[sub]);
            o[sub] = wmma_f16(ap1.v, bv1, o[sub]);
        }
    }

    // Normalize and write merged-head layout [B, N, INNER] (e = h*64 + d)
#pragma unroll
    for (int sub = 0; sub < 4; ++sub) {
#pragma unroll
        for (int r = 0; r < 8; ++r) {
            const int n = mBase + wave * 16 + hi * 8 + r;
            const int e = h * 64 + sub * 16 + lm;
            attO[((size_t)b * N_ + n) * INNER_ + e] = (half_t)(o[sub][r] / lrow[r]);
        }
    }
}

// ---------------------------------------------------------------------------
// Kernel 3: output projection. out[b,n,c] = sum_e attO[b,n,e] * Wo[c,e]
// grid (N/64, DIM/64, B), block 128. A-tile staged with async copy to LDS
// (GLOBAL_LOAD_ASYNC_TO_LDS_B128, ASYNCcnt-tracked); B-tile converts f32->f16.
// ---------------------------------------------------------------------------
__global__ __launch_bounds__(128)
void out_proj_kernel(const half_t* __restrict__ attO, const float* __restrict__ Wo,
                     float* __restrict__ out) {
    __shared__ __attribute__((aligned(32))) half_t At[64 * 64];
    __shared__ __attribute__((aligned(32))) half_t Bt[64 * 64];

    const int t    = threadIdx.x;
    const int wave = t >> 5, lane = t & 31, hi = lane >> 4, lm = lane & 15;
    const int b = blockIdx.z;
    const int nBase = blockIdx.x * 64;
    const int cBase = blockIdx.y * 64;

    v8f acc[4];
#pragma unroll
    for (int s = 0; s < 4; ++s) acc[s] = {};

    for (int kb = 0; kb < INNER_; kb += 64) {
        __syncthreads();
        // A tile: f16 workspace -> LDS via async DMA (no VGPR round trip)
#pragma unroll
        for (int i = 0; i < 4; ++i) {
            const int u   = t + i * 128;          // 0..511
            const int row = u >> 3;               // 0..63
            const int c8  = (u & 7) * 8;          // halves, 16B chunks
            const unsigned lds = (unsigned)(uintptr_t)&At[row * 64 + c8];
            const unsigned long long ga = (unsigned long long)
                (attO + ((size_t)b * N_ + nBase + row) * INNER_ + kb + c8);
            asm volatile("global_load_async_to_lds_b128 %0, %1, off"
                         :: "v"(lds), "v"(ga) : "memory");
        }
        // B tile: Wo f32 -> f16, k-contiguous per output column
#pragma unroll
        for (int i = 0; i < 8; ++i) {
            const int idx = t + i * 128;
            const int row = idx >> 4;
            const int c4  = (idx & 15) * 4;
            const float4 wb =
                *(const float4*)(Wo + (size_t)(cBase + row) * INNER_ + kb + c4);
            v4h hb; hb[0] = (half_t)wb.x; hb[1] = (half_t)wb.y;
                    hb[2] = (half_t)wb.z; hb[3] = (half_t)wb.w;
            *(v4h*)&Bt[row * 64 + c4] = hb;
        }
        asm volatile("s_wait_asynccnt 0x0" ::: "memory");
        __syncthreads();

        AV a0, a1;
        a0.h8[0] = *(const v8h*)&At[(wave * 16 + lm) * 64 + hi * 16];
        a0.h8[1] = *(const v8h*)&At[(wave * 16 + lm) * 64 + 8 + hi * 16];
        a1.h8[0] = *(const v8h*)&At[(wave * 16 + lm) * 64 + 32 + hi * 16];
        a1.h8[1] = *(const v8h*)&At[(wave * 16 + lm) * 64 + 40 + hi * 16];
#pragma unroll
        for (int sub = 0; sub < 4; ++sub) {
            const v16h bf0 = *(const v16h*)&Bt[(sub * 16 + lm) * 64 + hi * 16];
            const v16h bf1 = *(const v16h*)&Bt[(sub * 16 + lm) * 64 + 32 + hi * 16];
            acc[sub] = wmma_f16(a0.v, bf0, acc[sub]);
            acc[sub] = wmma_f16(a1.v, bf1, acc[sub]);
        }
    }

#pragma unroll
    for (int sub = 0; sub < 4; ++sub) {
#pragma unroll
        for (int r = 0; r < 8; ++r) {
            const int m = nBase + wave * 16 + hi * 8 + r;
            const int c = cBase + sub * 16 + lm;
            out[((size_t)b * N_ + m) * DIM_ + c] = acc[sub][r];
        }
    }
}

// ---------------------------------------------------------------------------
extern "C" void kernel_launch(void* const* d_in, const int* in_sizes, int n_in,
                              void* d_out, int out_size, void* d_ws, size_t ws_size,
                              hipStream_t stream) {
    (void)in_sizes; (void)n_in; (void)out_size; (void)ws_size;
    const float* q    = (const float*)d_in[0];
    const float* k    = (const float*)d_in[1];
    const float* v    = (const float*)d_in[2];
    const float* bias = (const float*)d_in[3];
    const float* Wq   = (const float*)d_in[4];
    const float* Wk   = (const float*)d_in[5];
    const float* Wv   = (const float*)d_in[6];
    const float* Wo   = (const float*)d_in[7];
    float* out = (float*)d_out;

    const size_t QKV = (size_t)B_ * HEADS_ * N_ * DH_;  // 8,388,608 elems
    half_t* qh   = (half_t*)d_ws;
    half_t* kh   = qh + QKV;
    half_t* vh   = kh + QKV;
    half_t* attO = vh + QKV;                            // total 64 MB f16

    const dim3 blk(128);
    proj_qkv_kernel<<<dim3(N_ / 64, INNER_ / 64, B_ * 3), blk, 0, stream>>>(
        q, k, v, Wq, Wk, Wv, qh, kh, vh);
    flash_attn_kernel<<<dim3(N_ / 64, HEADS_, B_), blk, 0, stream>>>(
        qh, kh, vh, bias, attO);
    out_proj_kernel<<<dim3(N_ / 64, DIM_ / 64, B_), blk, 0, stream>>>(
        attO, Wo, out);
}
